// GCNResidualModel_73632919323005
// MI455X (gfx1250) — matrix operand order repaired and checked by reading, not compiled, not run on hardware
//
#include <hip/hip_runtime.h>
#include <hip/hip_bf16.h>
#include <stdint.h>

typedef _Float16 v8h  __attribute__((ext_vector_type(8)));
typedef _Float16 v16h __attribute__((ext_vector_type(16)));
typedef float    v4f  __attribute__((ext_vector_type(4)));
typedef float    v8f  __attribute__((ext_vector_type(8)));
typedef unsigned int u32x4 __attribute__((ext_vector_type(4)));
typedef int          i32x4 __attribute__((ext_vector_type(4)));
typedef int          i32x8 __attribute__((ext_vector_type(8)));

#define GEMM_BM 128
#define GEMM_BN 128
#define GEMM_BK 32

#if defined(__has_builtin)
#if __has_builtin(__builtin_amdgcn_tensor_load_to_lds)
#define HAVE_TDM 1
#endif
#endif
#if __has_include(<hip/amd_detail/amd_gfx1250_TDM.h>)
#define TDM_SIX_ARGS 1   // therock/clang-23 lane: 6-arg builtin
#endif

// ---------------------------------------------------------------------------
// TDM: DMA a 2-D f16 tile (tile1 rows x tile0 elems) from global into LDS.
// Rows beyond tensor_dim1 (and elems beyond tensor_dim0) are zero-filled by HW.
// LDS receives the tile packed row-major: [tile1][tile0] f16.
// ---------------------------------------------------------------------------
#if defined(HAVE_TDM)
__device__ __forceinline__ void tdm_load_tile_f16(
    unsigned lds_byte_off, const _Float16* gsrc,
    unsigned tensor_d0, unsigned tensor_d1, unsigned stride0_elems,
    unsigned tile_d0, unsigned tile_d1)
{
    unsigned long long ga = (unsigned long long)(uintptr_t)gsrc;
    u32x4 g0;
    g0[0] = 1u;                                           // count=1, user D#
    g0[1] = lds_byte_off;                                 // lds_addr
    g0[2] = (unsigned)(ga & 0xffffffffu);                 // global_addr[31:0]
    g0[3] = (unsigned)((ga >> 32) & 0x01ffffffu)          // global_addr[56:32]
            | (2u << 30);                                 // type = 2 ("image")
    i32x8 g1;
    g1[0] = (int)(1u << 16);                              // wg_mask=0, data_size=1 (2B)
    g1[1] = (int)((tensor_d0 & 0xffffu) << 16);           // dim0[15:0] at bits 63:48
    g1[2] = (int)((tensor_d0 >> 16) | ((tensor_d1 & 0xffffu) << 16));
    g1[3] = (int)((tensor_d1 >> 16) | (tile_d0 << 16));   // dim1 hi | tile_dim0
    g1[4] = (int)(tile_d1 & 0xffffu);                     // tile_dim1, tile_dim2=0
    g1[5] = (int)stride0_elems;                           // tensor_dim0_stride[31:0]
    g1[6] = 0;                                            // stride0 hi | stride1 lo
    g1[7] = 0;
    i32x4 z4 = {0, 0, 0, 0};
#if defined(TDM_SIX_ARGS)
    i32x8 z8 = {0, 0, 0, 0, 0, 0, 0, 0};
    __builtin_amdgcn_tensor_load_to_lds(g0, g1, z4, z4, z8, 0);
#else
    __builtin_amdgcn_tensor_load_to_lds(g0, g1, z4, z4, 0);
#endif
}
#endif

// ---------------------------------------------------------------------------
// Degree / normalization (deg = 1 self-loop + in-degree; dis = rsqrt(deg))
// ---------------------------------------------------------------------------
__global__ void deg_init_kernel(float* __restrict__ degF, float* __restrict__ degB, int N) {
    int i = blockIdx.x * blockDim.x + threadIdx.x;
    if (i < N) { degF[i] = 1.0f; degB[i] = 1.0f; }
}

__global__ void deg_count_kernel(float* __restrict__ deg, const long long* __restrict__ cols,
                                 long long E) {
    long long e = (long long)blockIdx.x * blockDim.x + threadIdx.x;
    if (e < E) atomicAdd(&deg[(int)cols[e]], 1.0f);
}

__global__ void deg_finalize_kernel(float* __restrict__ degF, float* __restrict__ degB, int N) {
    int i = blockIdx.x * blockDim.x + threadIdx.x;
    if (i < N) { degF[i] = rsqrtf(degF[i]); degB[i] = rsqrtf(degB[i]); }
}

// ---------------------------------------------------------------------------
// f32 -> f16 conversion (activations), vectorized x4
// ---------------------------------------------------------------------------
__global__ void convert_f16_kernel(const float* __restrict__ X, _Float16* __restrict__ Xh,
                                   long long total4) {
    long long idx = (long long)blockIdx.x * blockDim.x + threadIdx.x;
    if (idx >= total4) return;
    v4f v = *(const v4f*)(X + idx * 4);
    _Float16* o = Xh + idx * 4;
    o[0] = (_Float16)v[0]; o[1] = (_Float16)v[1];
    o[2] = (_Float16)v[2]; o[3] = (_Float16)v[3];
}

// Weight transpose+convert: W[K x M] f32 row-major -> Wt[M x K] f16
__global__ void convert_wt_kernel(const float* __restrict__ W, _Float16* __restrict__ Wt,
                                  int K, int M) {
    int idx = blockIdx.x * blockDim.x + threadIdx.x;
    if (idx >= K * M) return;
    int o = idx / K;
    int k = idx - o * K;
    Wt[idx] = (_Float16)W[(size_t)k * M + o];
}

// ---------------------------------------------------------------------------
// WMMA GEMM (f16 in, f32 out): C[N,M] = A[N,K] @ Bt[M,K]^T (+bias) (+=C)
// A row-major f16, Bt = transposed weights [M][K] f16.
// A/B tiles DMA'd into LDS by the Tensor Data Mover (wave 0), zero VGPR bounce.
// flags bit0 = add bias, bit1 = accumulate onto C.
// ---------------------------------------------------------------------------
__global__ __launch_bounds__(256)
void wmma_gemm_f16_kernel(const _Float16* __restrict__ A, const _Float16* __restrict__ Bt,
                          const float* __restrict__ bias, float* __restrict__ C,
                          int N, int K, int M, int flags)
{
    __shared__ __align__(16) _Float16 As[GEMM_BM * GEMM_BK];  // [row][k], stride 32
    __shared__ __align__(16) _Float16 Bs[GEMM_BN * GEMM_BK];  // [col][k], stride 32

    const int tid     = threadIdx.x;
    const int lane    = tid & 31;
    const int wid     = tid >> 5;
    const int l15     = lane & 15;
    const int halfsel = lane >> 4;                 // 0: lanes 0-15, 1: lanes 16-31

    const int nBlocksM = (N + GEMM_BM - 1) / GEMM_BM;
    const int rowBlock = (blockIdx.x % nBlocksM) * GEMM_BM;
    const int colBlock = (blockIdx.x / nBlocksM) * GEMM_BN;

    const int waveRowBase = (wid >> 2) * 64;       // 0 / 64
    const int waveColBase = (wid & 3) * 32;        // 0 / 32 / 64 / 96

    v8f acc[4][2];
    for (int m = 0; m < 4; ++m)
        for (int n = 0; n < 2; ++n)
            for (int i = 0; i < 8; ++i) acc[m][n][i] = 0.0f;

    for (int k0 = 0; k0 < K; k0 += GEMM_BK) {
        __syncthreads();   // previous iteration's readers done before overwrite
#if defined(HAVE_TDM)
        if (wid == 0) {
            // A tile: rows [rowBlock, rowBlock+128) x k [k0, k0+32); HW zero-fills OOB rows
            tdm_load_tile_f16((unsigned)(uintptr_t)(void*)As,
                              A + (size_t)rowBlock * K + k0,
                              (unsigned)K, (unsigned)(N - rowBlock), (unsigned)K,
                              GEMM_BK, GEMM_BM);
            // B tile: cols [colBlock, colBlock+128) x k [k0, k0+32) from Wt[M][K]
            tdm_load_tile_f16((unsigned)(uintptr_t)(void*)Bs,
                              Bt + (size_t)colBlock * K + k0,
                              (unsigned)K, (unsigned)(M - colBlock), (unsigned)K,
                              GEMM_BK, GEMM_BN);
            __builtin_amdgcn_s_wait_tensorcnt(0);
        }
#else
        // Fallback: 16B-per-thread staged copies (2 chunks per matrix per thread)
        #pragma unroll
        for (int p = 0; p < 2; ++p) {
            int linear = p * 256 + tid;            // 0..511 (128 rows * 4 chunks)
            int r  = linear >> 2;
            int c8 = (linear & 3) * 8;
            int grow = rowBlock + r;
            v8h va;
            if (grow < N) va = *(const v8h*)(A + (size_t)grow * K + k0 + c8);
            else for (int i = 0; i < 8; ++i) va[i] = (_Float16)0.0f;
            *(v8h*)(As + r * GEMM_BK + c8) = va;
            v8h vb = *(const v8h*)(Bt + (size_t)(colBlock + r) * K + k0 + c8);
            *(v8h*)(Bs + r * GEMM_BK + c8) = vb;
        }
#endif
        __syncthreads();

        // B fragments: lane holds col = base+l15, 16 contiguous K values
        v16h bfrag[2];
        #pragma unroll
        for (int n = 0; n < 2; ++n)
            bfrag[n] = *(const v16h*)(Bs + (waveColBase + n * 16 + l15) * GEMM_BK
                                         + halfsel * 16);
        // A fragments + 8 WMMAs
        #pragma unroll
        for (int m = 0; m < 4; ++m) {
            const _Float16* ap = As + (waveRowBase + m * 16 + l15) * GEMM_BK + halfsel * 16;
            v8h lo = *(const v8h*)ap;              // K = h*16 .. h*16+7
            v8h hi = *(const v8h*)(ap + 8);        // K = h*16+8 .. h*16+15
            v16h af;
            #pragma unroll
            for (int i = 0; i < 8; ++i) { af[i] = lo[i]; af[8 + i] = hi[i]; }
            #pragma unroll
            for (int n = 0; n < 2; ++n)
                acc[m][n] = __builtin_amdgcn_wmma_f32_16x16x32_f16(
                    false, af, false, bfrag[n], (short)0, acc[m][n], false, false);
        }
    }

    // epilogue: VGPR r holds M=r (lanes 0-15) / M=r+8 (lanes 16-31), N = l15
    #pragma unroll
    for (int m = 0; m < 4; ++m) {
        #pragma unroll
        for (int n = 0; n < 2; ++n) {
            int col = colBlock + waveColBase + n * 16 + l15;
            float bv = (flags & 1) ? bias[col] : 0.0f;
            #pragma unroll
            for (int r = 0; r < 8; ++r) {
                int row = rowBlock + waveRowBase + m * 16 + r + halfsel * 8;
                if (row < N) {
                    float v = acc[m][n][r] + bv;
                    size_t idx = (size_t)row * M + col;
                    if (flags & 2) v += C[idx];
                    C[idx] = v;
                }
            }
        }
    }
}

// ---------------------------------------------------------------------------
// out[i,:] = h[i,:] * dis[i]^2 + bias   (self-loop term + conv bias), D = 256
// ---------------------------------------------------------------------------
__global__ void selfloop_kernel(float* __restrict__ out, const float* __restrict__ h,
                                const float* __restrict__ dis, const float* __restrict__ bias,
                                long long total) {
    long long idx = (long long)blockIdx.x * blockDim.x + threadIdx.x;
    if (idx >= total) return;
    int row = (int)(idx >> 8);
    int col = (int)(idx & 255);
    float d = dis[row];
    out[idx] = h[idx] * d * d + bias[col];
}

// ---------------------------------------------------------------------------
// Edge scatter: out[col] += h[row] * (dis[row]*dis[col]); 64 lanes per edge.
// ---------------------------------------------------------------------------
__global__ __launch_bounds__(256)
void scatter_kernel(float* __restrict__ out, const float* __restrict__ h,
                    const long long* __restrict__ rows, const long long* __restrict__ cols,
                    const float* __restrict__ dis, long long E)
{
    long long e = (long long)blockIdx.x * 4 + (threadIdx.x >> 6);
    if (e >= E) return;
    __builtin_prefetch(rows + e + 4096, 0, 1);     // global_prefetch_b8 on edge list
    int r  = (int)rows[e];
    int cl = (int)cols[e];
    float norm = dis[r] * dis[cl];
    int c = (threadIdx.x & 63) * 4;
    v4f m = *(const v4f*)(h + (size_t)r * 256 + c);
    float* o = out + (size_t)cl * 256 + c;
    atomicAdd(o + 0, m[0] * norm);
    atomicAdd(o + 1, m[1] * norm);
    atomicAdd(o + 2, m[2] * norm);
    atomicAdd(o + 3, m[3] * norm);
}

// ---------------------------------------------------------------------------
// BatchNorm pipeline (training-mode stats over nodes), D = 256
// ---------------------------------------------------------------------------
__global__ void zero_stats_kernel(float* __restrict__ stats) {
    int t = blockIdx.x * blockDim.x + threadIdx.x;
    if (t < 512) stats[t] = 0.0f;
}

__global__ __launch_bounds__(256)
void relu_stats_kernel(float* __restrict__ h, float* __restrict__ stats, int N) {
    int ch   = threadIdx.x;                        // 256 channels == blockDim.x
    int row0 = blockIdx.x * 256;
    int row1 = row0 + 256; if (row1 > N) row1 = N;
    float s = 0.0f, s2 = 0.0f;
    for (int r = row0; r < row1; ++r) {
        size_t idx = (size_t)r * 256 + ch;
        float v = h[idx];
        v = fmaxf(v, 0.0f);
        h[idx] = v;
        s += v; s2 += v * v;
    }
    atomicAdd(&stats[ch], s);
    atomicAdd(&stats[256 + ch], s2);
}

__global__ void bn_finalize_kernel(float* __restrict__ stats, const float* __restrict__ gamma,
                                   const float* __restrict__ beta, int N) {
    int ch = threadIdx.x;
    if (ch >= 256) return;
    float inv_n = 1.0f / (float)N;
    float mean  = stats[ch] * inv_n;
    float var   = stats[256 + ch] * inv_n - mean * mean;   // biased variance
    float sc    = gamma[ch] * rsqrtf(var + 1e-5f);
    stats[512 + ch] = sc;
    stats[768 + ch] = beta[ch] - mean * sc;
}

__global__ void bn_residual_kernel(float* __restrict__ out, const float* __restrict__ h,
                                   const float* __restrict__ res, const float* __restrict__ stats,
                                   long long total) {
    long long idx = (long long)blockIdx.x * blockDim.x + threadIdx.x;
    if (idx >= total) return;
    int ch = (int)(idx & 255);
    out[idx] = h[idx] * stats[512 + ch] + stats[768 + ch] + res[idx];
}

// ---------------------------------------------------------------------------
// Host-side orchestration
// ---------------------------------------------------------------------------
static inline void launch_gemm(const _Float16* A, const _Float16* Bt, const float* bias,
                               float* C, int N, int K, int M, int flags, hipStream_t stream) {
    int nBlocksM = (N + GEMM_BM - 1) / GEMM_BM;
    int grid = nBlocksM * (M / GEMM_BN);
    wmma_gemm_f16_kernel<<<grid, 256, 0, stream>>>(A, Bt, bias, C, N, K, M, flags);
}

extern "C" void kernel_launch(void* const* d_in, const int* in_sizes, int n_in,
                              void* d_out, int out_size, void* d_ws, size_t ws_size,
                              hipStream_t stream) {
    const int   N = in_sizes[0] / 128;             // 100000
    const long long E = in_sizes[1] / 2;           // 3.2M
    const float*     x0     = (const float*)d_in[0];
    const long long* fwd_ei = (const long long*)d_in[1];   // [2, E] int64
    const long long* bwd_ei = (const long long*)d_in[2];
    // d_in[3] = edge_attr (unused by reference forward)

    auto P = [&](int layer, int i) -> const float* {
        return (const float*)d_in[4 + layer * 10 + i];
    };

    // workspace layout
    float* ws    = (float*)d_ws;
    float* disF  = ws;
    float* disB  = ws + N;
    float* stats = ws + 2 * (size_t)N;             // 1024: sum|sumsq|scale|shift
    float* bufA  = stats + 1024;
    float* bufB  = bufA + (size_t)N * 256;
    float* bufC  = bufB + (size_t)N * 256;
    float* bufD  = bufC + (size_t)N * 256;
    _Float16* hX = (_Float16*)(bufD + (size_t)N * 256);    // f16 features
    _Float16* hY = hX + (size_t)N * 256;
    _Float16* wt = hY + (size_t)N * 256;                   // transposed f16 weights (<=128KB)

    const long long totalD = (long long)N * 256;
    const int eltGrid     = (int)((totalD + 255) / 256);
    const int cvtGridD    = (int)((totalD / 4 + 255) / 256);
    const int nGrid       = (N + 255) / 256;
    const int edgeGrid    = (int)((E + 255) / 256);
    const int scatterGrid = (int)((E + 3) / 4);

    // ---- degree normalization (edge-structure only; computed once) ----
    deg_init_kernel<<<nGrid, 256, 0, stream>>>(disF, disB, N);
    deg_count_kernel<<<edgeGrid, 256, 0, stream>>>(disF, fwd_ei + E, E);
    deg_count_kernel<<<edgeGrid, 256, 0, stream>>>(disB, bwd_ei + E, E);
    deg_finalize_kernel<<<nGrid, 256, 0, stream>>>(disF, disB, N);

    const float* xcur = x0;
    for (int l = 0; l < 3; ++l) {
        const int K = (l == 0) ? 128 : 256;
        const long long totalK = (long long)N * K;
        const int wtGrid = (K * 256 + 255) / 256;

        // f16 activations for this layer's input
        convert_f16_kernel<<<(int)((totalK / 4 + 255) / 256), 256, 0, stream>>>(xcur, hX, totalK / 4);

        // h = x @ W via WMMA (conv bias folded into self-loop init)
        convert_wt_kernel<<<wtGrid, 256, 0, stream>>>(P(l, 0), wt, K, 256);
        launch_gemm(hX, wt, nullptr, bufA, N, K, 256, 0, stream);
        convert_wt_kernel<<<wtGrid, 256, 0, stream>>>(P(l, 2), wt, K, 256);
        launch_gemm(hX, wt, nullptr, bufB, N, K, 256, 0, stream);

        // residual projection (layer 0 only) -> d_out; aliases bn_residual dest safely
        const float* res;
        if (l == 0) {
            convert_wt_kernel<<<wtGrid, 256, 0, stream>>>(P(l, 8), wt, K, 256);
            launch_gemm(hX, wt, P(l, 9), (float*)d_out, N, K, 256, /*bias*/1, stream);
            res = (const float*)d_out;
        } else {
            res = xcur;
        }

        // conv aggregation: self-loop term, then edge scatter-add (atomic f32)
        selfloop_kernel<<<eltGrid, 256, 0, stream>>>(bufC, bufA, disF, P(l, 1), totalD);
        scatter_kernel<<<scatterGrid, 256, 0, stream>>>(bufC, bufA, fwd_ei, fwd_ei + E, disF, E);
        selfloop_kernel<<<eltGrid, 256, 0, stream>>>(bufD, bufB, disB, P(l, 3), totalD);
        scatter_kernel<<<scatterGrid, 256, 0, stream>>>(bufD, bufB, bwd_ei, bwd_ei + E, disB, E);

        // merge: concat([fwd,bwd]) @ Wm == fwd @ Wm_top + bwd @ Wm_bot
        convert_f16_kernel<<<cvtGridD, 256, 0, stream>>>(bufC, hX, totalD / 4);
        convert_f16_kernel<<<cvtGridD, 256, 0, stream>>>(bufD, hY, totalD / 4);
        convert_wt_kernel<<<256, 256, 0, stream>>>(P(l, 4), wt, 256, 256);
        launch_gemm(hX, wt, P(l, 5), bufA, N, 256, 256, /*bias*/1, stream);
        convert_wt_kernel<<<256, 256, 0, stream>>>(P(l, 4) + 256 * 256, wt, 256, 256);
        launch_gemm(hY, wt, nullptr, bufA, N, 256, 256, /*accum*/2, stream);

        // relu -> batchnorm stats -> fused scale/shift + residual
        zero_stats_kernel<<<2, 256, 0, stream>>>(stats);
        relu_stats_kernel<<<nGrid, 256, 0, stream>>>(bufA, stats, N);
        bn_finalize_kernel<<<1, 256, 0, stream>>>(stats, P(l, 6), P(l, 7), N);

        float* dest = (l == 0) ? (float*)d_out : (l == 1) ? bufC : (float*)d_out;
        bn_residual_kernel<<<eltGrid, 256, 0, stream>>>(dest, bufA, res, stats, totalD);
        xcur = dest;
    }
}
